// GraphAttentionLayer_64415919505491
// MI455X (gfx1250) — compile-verified, matrix operand order checked by reading, not run on hardware
//
#include <hip/hip_runtime.h>
#include <hip/hip_bf16.h>

#define NUM_HEADS   4
#define OUTPUT_DIM  32
#define NUM_NODES   50000
#define NUM_EDGES   800000
#define INPUT_DIM   128
#define PROJ_DIM    (NUM_HEADS * OUTPUT_DIM)   // 128

typedef __attribute__((ext_vector_type(16))) _Float16 v16h;
typedef __attribute__((ext_vector_type(8)))  _Float16 v8h;
typedef __attribute__((ext_vector_type(8)))  float    v8f;

// ---------------------------------------------------------------------------
// Kernel 0: convert W (128x128 f32, row-major) to f16, pre-swizzled into the
// exact WMMA B-fragment layout (32x16 tile per (coltile t, kstep k)):
//   half i of lane L belongs to row K = k0 + 2*((i/2)&3) + (i&1) + 16*(i/8)
//   + 8*(L/16), column N = 16*t + (L&15).
// Packed so each lane's 16 halves are 32 contiguous bytes -> one b128 pair.
// 1024 threads = one thread per (t,k,L).
// ---------------------------------------------------------------------------
__global__ __launch_bounds__(1024)
void pack_W_kernel(const float* __restrict__ W, _Float16* __restrict__ packedW) {
  int tid = threadIdx.x;                 // 0..1023
  int L   = tid & 31;
  int k   = (tid >> 5) & 3;              // k-step 0..3
  int t   = tid >> 7;                    // column tile 0..7
  int col = t * 16 + (L & 15);
  int k0  = k * 32;
  int hiK = (L >> 4) * 8;
  _Float16* dst = packedW + (size_t)tid * 16;
  #pragma unroll
  for (int i = 0; i < 16; ++i) {
    int Kl = 2 * ((i >> 1) & 3) + (i & 1) + 16 * (i >> 3) + hiK;
    dst[i] = (_Float16)W[(size_t)(k0 + Kl) * PROJ_DIM + col];
  }
}

// ---------------------------------------------------------------------------
// Kernel 1: proj = X @ W via v_wmma_f32_16x16x32_f16.
// One block (256 thr = 8 wave32) per 16-row tile; wave w owns columns
// [16w, 16w+16). A tile staged f32->f16 through LDS; 4 WMMAs over K=128.
// ---------------------------------------------------------------------------
__global__ __launch_bounds__(256)
void wmma_proj_kernel(const float* __restrict__ X,
                      const _Float16* __restrict__ packedW,
                      float* __restrict__ proj) {
  __shared__ _Float16 ldsA[16 * INPUT_DIM];   // 4 KB

  const int tid  = threadIdx.x;
  const size_t row0 = (size_t)blockIdx.x * 16;

  // Stage 16x128 f32 -> f16 into LDS (each thread converts 8 elements).
  {
    const float* src = X + row0 * INPUT_DIM + (size_t)tid * 8;
    _Float16* d = ldsA + tid * 8;
    #pragma unroll
    for (int j = 0; j < 8; ++j) d[j] = (_Float16)src[j];
  }
  __syncthreads();

  const int wave   = tid >> 5;
  const int lane   = tid & 31;
  const int laneHi = lane >> 4;
  const int n0     = wave * 16;

  const _Float16* rowp = ldsA + (size_t)(lane & 15) * INPUT_DIM;  // A row M=lane%16
  const _Float16* bp   = packedW + ((size_t)wave * 4 * 32 + lane) * 16;

  v8f c = {};
  #pragma unroll
  for (int k = 0; k < 4; ++k) {
    const int o1 = k * 32 + laneHi * 8;
    v8h alo = *(const v8h*)(rowp + o1);          // K = o1 .. o1+7
    v8h ahi = *(const v8h*)(rowp + o1 + 16);     // K = o1+16 .. o1+23
    v16h a  = __builtin_shufflevector(alo, ahi,
                0,1,2,3,4,5,6,7,8,9,10,11,12,13,14,15);
    v16h b  = *(const v16h*)(bp + (size_t)k * 32 * 16);
    c = __builtin_amdgcn_wmma_f32_16x16x32_f16(
          /*neg_a=*/false, a, /*neg_b=*/false, b,
          /*c_mod=*/(short)0, c, /*reuse_a=*/false, /*reuse_b=*/false);
  }

  // C layout: lane holds col N=lane%16, rows M = r + 8*(lane/16), r=0..7.
  float* outp = proj + (row0 + 8 * laneHi) * PROJ_DIM + n0 + (lane & 15);
  #pragma unroll
  for (int r = 0; r < 8; ++r) outp[(size_t)r * PROJ_DIM] = c[r];
}

// ---------------------------------------------------------------------------
// Kernel 2: zero the output accumulator (harness poisons it with 0xAA).
// ---------------------------------------------------------------------------
__global__ void zero_out_kernel(float4* __restrict__ out, int n4) {
  int i = blockIdx.x * blockDim.x + threadIdx.x;
  if (i < n4) out[i] = make_float4(0.f, 0.f, 0.f, 0.f);
}

// ---------------------------------------------------------------------------
// Kernel 3: per-node attention pre-dots.
// s_src[n,h] = <proj[n,h,:], att[:32]>, s_tgt[n,h] = <proj[n,h,:], att[32:]>.
// One thread per (node, head).
// ---------------------------------------------------------------------------
__global__ __launch_bounds__(256)
void node_scores_kernel(const float* __restrict__ proj,
                        const float* __restrict__ att,
                        float* __restrict__ s_src,
                        float* __restrict__ s_tgt) {
  int gid = blockIdx.x * blockDim.x + threadIdx.x;
  if (gid >= NUM_NODES * NUM_HEADS) return;
  int n = gid >> 2, h = gid & 3;
  const float* p = proj + (size_t)n * PROJ_DIM + h * OUTPUT_DIM;
  float ss = 0.f, st = 0.f;
  #pragma unroll
  for (int d = 0; d < OUTPUT_DIM; ++d) {
    float v = p[d];
    ss = fmaf(v, att[d], ss);
    st = fmaf(v, att[OUTPUT_DIM + d], st);
  }
  s_src[gid] = ss;
  s_tgt[gid] = st;
}

// ---------------------------------------------------------------------------
// Kernel 4: edge phase. One wave per edge (strided loop).
// Lane d handles output dim d: 4 coalesced 128B gathers of proj[tgt] and one
// global_atomic_add_f32 into out[src*32+d]. Scores/softmax are wave-uniform.
// proj (25.6 MB) and out (6.4 MB) are L2-resident on the 192 MB L2.
// ---------------------------------------------------------------------------
__global__ __launch_bounds__(256)
void edge_kernel(const long long* __restrict__ e_src,
                 const long long* __restrict__ e_tgt,
                 const float* __restrict__ proj,
                 const float* __restrict__ s_src,
                 const float* __restrict__ s_tgt,
                 float* __restrict__ out) {
  const int lane = threadIdx.x & 31;
  long long wid = ((long long)blockIdx.x * blockDim.x + threadIdx.x) >> 5;
  const long long nw = ((long long)gridDim.x * blockDim.x) >> 5;

  for (long long e = wid; e < NUM_EDGES; e += nw) {
    const long long s = e_src[e];
    const long long t = e_tgt[e];

    float sc[NUM_HEADS];
    float m = -1e30f;
    #pragma unroll
    for (int h = 0; h < NUM_HEADS; ++h) {
      float x = s_src[s * NUM_HEADS + h] + s_tgt[t * NUM_HEADS + h];
      x = (x > 0.f) ? x : 0.01f * x;       // leaky_relu(0.01)
      sc[h] = x;
      m = fmaxf(m, x);
    }
    float sum = 0.f;
    #pragma unroll
    for (int h = 0; h < NUM_HEADS; ++h) { sc[h] = __expf(sc[h] - m); sum += sc[h]; }
    const float inv = 0.25f / sum;         // alpha normalization * head-mean

    const float* tp = proj + (size_t)t * PROJ_DIM + lane;
    float acc = 0.f;
    #pragma unroll
    for (int h = 0; h < NUM_HEADS; ++h) acc = fmaf(sc[h], tp[h * OUTPUT_DIM], acc);

    atomicAdd(out + (size_t)s * OUTPUT_DIM + lane, acc * inv);
  }
}

// ---------------------------------------------------------------------------
extern "C" void kernel_launch(void* const* d_in, const int* in_sizes, int n_in,
                              void* d_out, int out_size, void* d_ws, size_t ws_size,
                              hipStream_t stream) {
  (void)in_sizes; (void)n_in; (void)out_size; (void)ws_size;

  const float*     X   = (const float*)d_in[0];       // (50000,128) f32
  const long long* E   = (const long long*)d_in[1];   // (2,800000) i64
  const float*     W   = (const float*)d_in[2];       // (128,128) f32
  const float*     att = (const float*)d_in[3];       // (64,) f32
  float*           out = (float*)d_out;               // (50000,32) f32

  // Workspace layout (16B-aligned):
  //   packedW : 32 KB  (8 tiles * 4 ksteps * 32 lanes * 16 f16)
  //   proj    : 25.6 MB (50000 x 128 f32)
  //   s_src   : 0.8 MB  (50000 x 4 f32)
  //   s_tgt   : 0.8 MB
  char* ws = (char*)d_ws;
  _Float16* packedW = (_Float16*)ws;
  float* proj  = (float*)(ws + 32768);
  float* s_src = (float*)(ws + 32768 + (size_t)NUM_NODES * PROJ_DIM * sizeof(float));
  float* s_tgt = s_src + (size_t)NUM_NODES * NUM_HEADS;

  pack_W_kernel<<<1, 1024, 0, stream>>>(W, packedW);

  wmma_proj_kernel<<<NUM_NODES / 16, 256, 0, stream>>>(X, packedW, proj);

  const int n4 = NUM_NODES * OUTPUT_DIM / 4;
  zero_out_kernel<<<(n4 + 255) / 256, 256, 0, stream>>>((float4*)out, n4);

  node_scores_kernel<<<(NUM_NODES * NUM_HEADS + 255) / 256, 256, 0, stream>>>(
      proj, att, s_src, s_tgt);

  edge_kernel<<<2048, 256, 0, stream>>>(E, E + NUM_EDGES, proj, s_src, s_tgt, out);
}
